// DeformConv_14113262535347
// MI455X (gfx1250) — matrix operand order repaired and checked by reading, not compiled, not run on hardware
//
#include <hip/hip_runtime.h>
#include <hip/hip_bf16.h>

// ---------------- problem constants ----------------
#define NN   4
#define CC   256
#define HH   96
#define WW   96
#define OO   256
#define HW   (HH*WW)          // 9216
#define KK   9
#define CKD  (CC*KK)          // 2304 reduction dim
#define STILE 64              // spatial tile per block
#define NSTILE (HW/STILE)     // 144 spatial tiles per image

typedef __attribute__((ext_vector_type(16))) __bf16 v16bf;
typedef __attribute__((ext_vector_type(8)))  float  v8f;

// fp32 -> bf16 round-to-nearest-even
__device__ __forceinline__ unsigned short f2bf(float f) {
    unsigned int u = __float_as_uint(f);
    u = (u + 0x7FFFu + ((u >> 16) & 1u)) >> 16;
    return (unsigned short)u;
}

// ---------------- kernel A: 3x3 offset conv (18 channels) ----------------
__global__ void offset_conv_kernel(const float* __restrict__ x,
                                   const float* __restrict__ w_off,
                                   const float* __restrict__ b_off,
                                   float* __restrict__ off) {
    int idx = blockIdx.x * 256 + threadIdx.x;
    if (idx >= NN * 18 * HW) return;
    int w  = idx % WW;
    int h  = (idx / WW) % HH;
    int ko = (idx / HW) % 18;        // uniform within a block (HW % 256 == 0)
    int n  = idx / (HW * 18);

    float s = b_off[ko];
    const float* xb = x + (size_t)n * CC * HW;
    const float* wb = w_off + (size_t)ko * CC * 9;
    for (int c = 0; c < CC; ++c) {
        const float* xc = xb + (size_t)c * HW;
        const float* wc = wb + c * 9;
#pragma unroll
        for (int ty = 0; ty < 3; ++ty) {
            int yy = h + ty - 1;
            if ((unsigned)yy < (unsigned)HH) {
#pragma unroll
                for (int tx = 0; tx < 3; ++tx) {
                    int xx = w + tx - 1;
                    if ((unsigned)xx < (unsigned)WW)
                        s += xc[yy * WW + xx] * wc[ty * 3 + tx];
                }
            }
        }
    }
    off[idx] = s;
}

// ---------------- kernel B: weights fp32 (O,C,3,3) -> bf16 [o][k][c] ----------------
__global__ void wconv_kernel(const float* __restrict__ w_dcn,
                             unsigned short* __restrict__ wbf) {
    int t = blockIdx.x * 256 + threadIdx.x;
    if (t >= OO * CKD) return;
    int o   = t / CKD;
    int rem = t % CKD;
    int k   = rem / CC;
    int c   = rem % CC;
    wbf[t] = f2bf(w_dcn[(size_t)o * CKD + c * 9 + k]);
}

// ---------------- kernel C: fused deform-sample + bf16 WMMA GEMM ----------------
// grid: NN*NSTILE blocks, 512 threads (16 waves). Each block: 64 spatial x 256 out-ch.
__global__ void __launch_bounds__(512)
deform_wmma_kernel(const float* __restrict__ x,
                   const float* __restrict__ off,
                   const unsigned short* __restrict__ wbf,
                   float* __restrict__ ybuf) {
    __shared__ unsigned short Bs[STILE * 32];   // [spatial 64][ck 32] bf16, 4 KB

    const int t      = threadIdx.x;
    const int n      = blockIdx.x / NSTILE;
    const int s_base = (blockIdx.x % NSTILE) * STILE;
    const int lane   = t & 31;
    const int wid    = t >> 5;                  // 0..15 -> o-tile
    const int row    = lane & 15;
    const int hi     = lane >> 4;
    const int o_base = wid * 16;

    v8f acc[4];
#pragma unroll
    for (int ss = 0; ss < 4; ++ss)
#pragma unroll
        for (int j = 0; j < 8; ++j) acc[ss][j] = 0.0f;

    // sampling-thread coordinates (fixed across chunks)
    const int s_l  = t >> 3;                    // 0..63 spatial within tile
    const int c4   = (t & 7) << 2;              // 0..28, 4 channels per thread
    const int sg   = s_base + s_l;
    const int h    = sg / WW;
    const int w    = sg - h * WW;
    const float* xb = x + (size_t)n * CC * HW;
    const float* offn = off + (size_t)n * 18 * HW;

    for (int kc = 0; kc < CKD / 32; ++kc) {     // 72 chunks
        const int k  = kc >> 3;                 // 0..8
        const int cb = (kc & 7) << 5;           // channel base 0..224

        // ---- build bf16 B tile in LDS: Bs[s][ck] = bilinear sample ----
        {
            const float dy = offn[(k * 2    ) * HW + sg];
            const float dx = offn[(k * 2 + 1) * HW + sg];
            const float py = (float)(h + (k / 3) - 1) + dy;
            const float px = (float)(w + (k % 3) - 1) + dx;
            const float y0f = floorf(py), x0f = floorf(px);
            const float wy = py - y0f,  wx = px - x0f;
            const int y0 = (int)y0f, x0 = (int)x0f;
            const int y1 = y0 + 1,   x1 = x0 + 1;
            const int y0c = min(max(y0, 0), HH - 1), y1c = min(max(y1, 0), HH - 1);
            const int x0c = min(max(x0, 0), WW - 1), x1c = min(max(x1, 0), WW - 1);
            const float my0 = ((unsigned)y0 < (unsigned)HH) ? 1.f : 0.f;
            const float my1 = ((unsigned)y1 < (unsigned)HH) ? 1.f : 0.f;
            const float mx0 = ((unsigned)x0 < (unsigned)WW) ? 1.f : 0.f;
            const float mx1 = ((unsigned)x1 < (unsigned)WW) ? 1.f : 0.f;
            const float w00 = (1.f - wy) * (1.f - wx) * my0 * mx0;
            const float w01 = (1.f - wy) * wx         * my0 * mx1;
            const float w10 = wy * (1.f - wx)         * my1 * mx0;
            const float w11 = wy * wx                 * my1 * mx1;
            const int i00 = y0c * WW + x0c, i01 = y0c * WW + x1c;
            const int i10 = y1c * WW + x0c, i11 = y1c * WW + x1c;

            const float* xc0 = xb + (size_t)(cb + c4) * HW;
            unsigned short hs[4];
#pragma unroll
            for (int cc = 0; cc < 4; ++cc) {
                const float* xc = xc0 + (size_t)cc * HW;
                float v = w00 * xc[i00] + w01 * xc[i01]
                        + w10 * xc[i10] + w11 * xc[i11];
                hs[cc] = f2bf(v);
            }
            uint2 pk;
            pk.x = (unsigned)hs[0] | ((unsigned)hs[1] << 16);
            pk.y = (unsigned)hs[2] | ((unsigned)hs[3] << 16);
            *(uint2*)&Bs[s_l * 32 + c4] = pk;
        }
        __syncthreads();

        // ---- A fragment: wbf[o_base+row][k*256 + cb + ...], ISA 16-bit A layout ----
        union { v16bf v; uint4 q[2]; } A;
        const unsigned short* ap =
            wbf + (size_t)(o_base + row) * CKD + k * CC + cb + hi * 8;
        A.q[0] = *(const uint4*)(ap);        // K 0..7   (hi=0) / 8..15  (hi=1)
        A.q[1] = *(const uint4*)(ap + 16);   // K 16..23 (hi=0) / 24..31 (hi=1)

#pragma unroll
        for (int ss = 0; ss < 4; ++ss) {
            union { v16bf v; uint4 q[2]; } B;
            const unsigned short* bp = &Bs[(ss * 16 + row) * 32 + hi * 8];
            B.q[0] = *(const uint4*)(bp);
            B.q[1] = *(const uint4*)(bp + 16);
            acc[ss] = __builtin_amdgcn_wmma_f32_16x16x32_bf16(
                false, A.v, false, B.v, (short)0, acc[ss], false, false);
        }
        __syncthreads();
    }

    // ---- store pre-BN y (C/D layout: VGPR j -> M = hi*8 + j, N = row) ----
    float* yb = ybuf + (size_t)n * OO * HW;
#pragma unroll
    for (int ss = 0; ss < 4; ++ss) {
        const int sgo = s_base + ss * 16 + row;
#pragma unroll
        for (int j = 0; j < 8; ++j) {
            const int o = o_base + hi * 8 + j;
            yb[(size_t)o * HW + sgo] = acc[ss][j];
        }
    }
}

// ---------------- kernel D: per-channel BN stats -> scale/shift ----------------
__global__ void bn_stats_kernel(const float* __restrict__ ybuf,
                                const float* __restrict__ gamma,
                                const float* __restrict__ beta,
                                float* __restrict__ scale,
                                float* __restrict__ shift) {
    const int o = blockIdx.x, tid = threadIdx.x;
    float s = 0.f, s2 = 0.f;
    for (int i = tid; i < NN * HW; i += 256) {
        int n = i / HW, sp = i % HW;
        float v = ybuf[((size_t)n * OO + o) * HW + sp];
        s += v; s2 += v * v;
    }
    __shared__ float sh[256], sh2[256];
    sh[tid] = s; sh2[tid] = s2;
    __syncthreads();
    for (int st = 128; st > 0; st >>= 1) {
        if (tid < st) { sh[tid] += sh[tid + st]; sh2[tid] += sh2[tid + st]; }
        __syncthreads();
    }
    if (tid == 0) {
        const float cnt = (float)(NN * HW);
        float mean = sh[0] / cnt;
        float var  = sh2[0] / cnt - mean * mean;
        float inv  = rsqrtf(var + 1e-5f);
        float sc   = gamma[o] * inv;
        scale[o] = sc;
        shift[o] = beta[o] - mean * sc;
    }
}

// ---------------- kernel E: apply BN + ReLU ----------------
__global__ void bn_apply_kernel(const float* __restrict__ ybuf,
                                const float* __restrict__ scale,
                                const float* __restrict__ shift,
                                float* __restrict__ out) {
    int idx = blockIdx.x * 256 + threadIdx.x;
    if (idx >= NN * OO * HW) return;
    int o = (idx / HW) % OO;
    float v = ybuf[idx] * scale[o] + shift[o];
    out[idx] = fmaxf(v, 0.f);
}

// ---------------- launcher ----------------
extern "C" void kernel_launch(void* const* d_in, const int* in_sizes, int n_in,
                              void* d_out, int out_size, void* d_ws, size_t ws_size,
                              hipStream_t stream) {
    const float* x     = (const float*)d_in[0];
    const float* w_off = (const float*)d_in[1];
    const float* b_off = (const float*)d_in[2];
    const float* w_dcn = (const float*)d_in[3];
    const float* gamma = (const float*)d_in[4];
    const float* beta  = (const float*)d_in[5];
    float* out = (float*)d_out;

    // workspace layout
    float*          off   = (float*)d_ws;                       // N*18*HW f32
    unsigned short* wbf   = (unsigned short*)(off + NN*18*HW);  // O*CKD bf16
    float*          ybuf  = (float*)(wbf + OO*CKD);             // N*O*HW f32
    float*          scale = ybuf + (size_t)NN*OO*HW;            // O
    float*          shift = scale + OO;                         // O

    // A: offset conv
    {
        int total = NN * 18 * HW;
        offset_conv_kernel<<<(total + 255) / 256, 256, 0, stream>>>(x, w_off, b_off, off);
    }
    // B: weight convert/reorder
    {
        int total = OO * CKD;
        wconv_kernel<<<(total + 255) / 256, 256, 0, stream>>>(w_dcn, wbf);
    }
    // C: fused deform-sample + WMMA GEMM
    deform_wmma_kernel<<<NN * NSTILE, 512, 0, stream>>>(x, off, wbf, ybuf);
    // D: BN stats
    bn_stats_kernel<<<OO, 256, 0, stream>>>(ybuf, gamma, beta, scale, shift);
    // E: BN apply + ReLU
    {
        int total = NN * OO * HW;
        bn_apply_kernel<<<(total + 255) / 256, 256, 0, stream>>>(ybuf, scale, shift, out);
    }
}
